// AttentionBlock_87737591922809
// MI455X (gfx1250) — matrix operand order repaired and checked by reading, not compiled, not run on hardware
//
#include <hip/hip_runtime.h>

typedef _Float16 half8 __attribute__((ext_vector_type(8)));
typedef _Float16 v16h  __attribute__((ext_vector_type(16)));
typedef float    v8f   __attribute__((ext_vector_type(8)));

#define B_    4
#define C_    512
#define N_    4096      // 64*64 pixels
#define M_    (B_ * N_) // 16384
#define G_    32

#define ASYNC_WAIT() asm volatile("s_wait_asynccnt 0" ::: "memory")

// ---------------------------------------------------------------------------
// WMMA fragment loaders (layouts per CDNA5 ISA 7.12.2, wave32)
// A 16x32 f16: lane L<16 -> row L, K {0..7,16..23}; lane>=16 -> row L-16, K {8..15,24..31}
// ---------------------------------------------------------------------------
__device__ __forceinline__ v16h load_a_frag(const _Float16* base, int ld, int lane) {
    int r  = lane & 15;
    int hi = (lane >> 4) << 3;
    const _Float16* p = base + r * ld + hi;
    half8 lo  = *(const half8*)(p);
    half8 hi8 = *(const half8*)(p + 16);
    return __builtin_shufflevector(lo, hi8, 0,1,2,3,4,5,6,7,8,9,10,11,12,13,14,15);
}

// B 32x16 f16 stored row-major-over-K (i.e. Bt[n][k]): lane L holds col N=L&15,
// K {0..7,16..23} for L<16, K {8..15,24..31} for L>=16
__device__ __forceinline__ v16h load_b_frag(const _Float16* base, int ld, int lane) {
    int n  = lane & 15;
    int kb = (lane >> 4) << 3;
    const _Float16* p = base + n * ld + kb;
    half8 lo = *(const half8*)(p);
    half8 hi = *(const half8*)(p + 16);
    return __builtin_shufflevector(lo, hi, 0,1,2,3,4,5,6,7,8,9,10,11,12,13,14,15);
}

// ---------------------------------------------------------------------------
// Block reductions (wave32)
// ---------------------------------------------------------------------------
__device__ __forceinline__ float block_sum(float v, float* sh) {
    #pragma unroll
    for (int m = 16; m >= 1; m >>= 1) v += __shfl_xor(v, m);
    if ((threadIdx.x & 31) == 0) sh[threadIdx.x >> 5] = v;
    __syncthreads();
    v = sh[0];
    #pragma unroll
    for (int i = 1; i < 8; ++i) v += sh[i];
    __syncthreads();
    return v;
}

__device__ __forceinline__ float block_max(float v, float* sh) {
    #pragma unroll
    for (int m = 16; m >= 1; m >>= 1) v = fmaxf(v, __shfl_xor(v, m));
    if ((threadIdx.x & 31) == 0) sh[threadIdx.x >> 5] = v;
    __syncthreads();
    v = sh[0];
    #pragma unroll
    for (int i = 1; i < 8; ++i) v = fmaxf(v, sh[i]);
    __syncthreads();
    return v;
}

// ---------------------------------------------------------------------------
// f32 -> f16 conversion
// ---------------------------------------------------------------------------
__global__ __launch_bounds__(256) void cvt_f16(const float* __restrict__ src,
                                               _Float16* __restrict__ dst, int n) {
    int i = blockIdx.x * 256 + threadIdx.x;
    if (i < n) dst[i] = (_Float16)src[i];
}

// ---------------------------------------------------------------------------
// GroupNorm: stats per (b, group) over contiguous 16*4096 f32 slab
// ---------------------------------------------------------------------------
__global__ __launch_bounds__(256) void gn_stats(const float* __restrict__ x,
                                                float* __restrict__ stats) {
    __shared__ float sh[8];
    int bg = blockIdx.x; // 0..127
    const float4* p = (const float4*)(x + (size_t)bg * 16 * N_);
    float s = 0.f, q = 0.f;
    for (int i = threadIdx.x; i < (16 * N_ / 4); i += 256) {
        float4 v = p[i];
        s += v.x + v.y + v.z + v.w;
        q += v.x * v.x + v.y * v.y + v.z * v.z + v.w * v.w;
    }
    s = block_sum(s, sh);
    q = block_sum(q, sh);
    if (threadIdx.x == 0) {
        const float inv = 1.f / (16.f * N_);
        float mean = s * inv;
        float var  = q * inv - mean * mean;
        stats[2 * bg]     = mean;
        stats[2 * bg + 1] = rsqrtf(var + 1e-5f);
    }
}

// Apply GN + transpose (b,c,n) f32 -> (b,n,c) f16
__global__ __launch_bounds__(256) void gn_apply(const float* __restrict__ x,
                                                const float* __restrict__ gw,
                                                const float* __restrict__ gb,
                                                const float* __restrict__ stats,
                                                _Float16* __restrict__ h16) {
    int bc = blockIdx.x;            // 0..2047
    int b = bc >> 9, c = bc & (C_ - 1);
    int g = c >> 4;
    float mean = stats[2 * (b * G_ + g)];
    float rs   = stats[2 * (b * G_ + g) + 1];
    float sc = rs * gw[c];
    float sf = gb[c] - mean * sc;
    const float* xp = x + (size_t)bc * N_;
    _Float16* hp = h16 + (size_t)b * N_ * C_ + c;
    for (int n = threadIdx.x; n < N_; n += 256)
        hp[(size_t)n * C_] = (_Float16)(xp[n] * sc + sf);
}

// ---------------------------------------------------------------------------
// Row softmax over 4096 f16 logits, in place
// ---------------------------------------------------------------------------
__global__ __launch_bounds__(256) void softmax_rows(_Float16* __restrict__ s) {
    __shared__ float sh[8];
    _Float16* p = s + (size_t)blockIdx.x * N_;
    float v[16];
    float mx = -3.0e38f;
    #pragma unroll
    for (int i = 0; i < 16; ++i) {
        v[i] = (float)p[threadIdx.x + i * 256];
        mx = fmaxf(mx, v[i]);
    }
    mx = block_max(mx, sh);
    float sum = 0.f;
    #pragma unroll
    for (int i = 0; i < 16; ++i) { v[i] = __expf(v[i] - mx); sum += v[i]; }
    sum = block_sum(sum, sh);
    float inv = 1.f / sum;
    #pragma unroll
    for (int i = 0; i < 16; ++i)
        p[threadIdx.x + i * 256] = (_Float16)(v[i] * inv);
}

// ---------------------------------------------------------------------------
// Tiled WMMA GEMM:  out[m][n] = sum_k A[m][k] * Bt[n][k]   (both row-major-over-K)
// Block tile 128x128, k-step 32, 8 waves (4 m-waves x 2 n-waves), wave tile 32x64.
// Double-buffered LDS with async global->LDS DMA (ASYNCcnt) per CDNA5 ISA ch.10:
// tile t+1 is in flight while WMMAs consume tile t; one s_wait_asynccnt + one
// barrier per k-step proves both "all waves done reading cur" and "all async
// writes to next landed".
// MODE 0: QKV split epilogue (+bias, v written transposed (c,n))
// MODE 1: f16 out, ld = Nn, scaled
// MODE 3: final: +bias +residual, f32 out in (b,c,n) layout
// ---------------------------------------------------------------------------
template <int MODE>
__global__ __launch_bounds__(256) void gemm_wmma(
    const _Float16* __restrict__ A, const _Float16* __restrict__ Bt,
    int Nn, int K,
    _Float16* __restrict__ oq, _Float16* __restrict__ ok, _Float16* __restrict__ ov,
    float* __restrict__ of32,
    const float* __restrict__ bias, const float* __restrict__ resid, float scale) {
    constexpr int BM = 128, BN = 128, BK = 32, LDT = 40;
    __shared__ alignas(16) _Float16 sA[2][BM * LDT];
    __shared__ alignas(16) _Float16 sB[2][BN * LDT];

    const int tid  = threadIdx.x;
    const int lane = tid & 31;
    const int wave = tid >> 5;
    const int wm = wave >> 1;   // 0..3  -> 32 rows each
    const int wn = wave & 1;    // 0..1  -> 64 cols each
    const int m0 = blockIdx.y * BM;
    const int n0 = blockIdx.x * BN;

    // issue async global->LDS copies for one 128x32 A tile + 128x32 B tile
    auto stage = [&](int kt, int buf) {
        #pragma unroll
        for (int it = 0; it < 2; ++it) {
            int idx = tid + it * 256;           // 0..511
            int row = idx >> 2;
            int c8  = (idx & 3) << 3;
            const _Float16* ga = &A[(size_t)(m0 + row) * K + kt + c8];
            const _Float16* gb = &Bt[(size_t)(n0 + row) * K + kt + c8];
            unsigned la = (unsigned)(size_t)(&sA[buf][row * LDT + c8]);
            unsigned lb = (unsigned)(size_t)(&sB[buf][row * LDT + c8]);
            asm volatile("global_load_async_to_lds_b128 %0, %1, off"
                         :: "v"(la), "v"(ga) : "memory");
            asm volatile("global_load_async_to_lds_b128 %0, %1, off"
                         :: "v"(lb), "v"(gb) : "memory");
        }
    };

    v8f acc[2][4];
    #pragma unroll
    for (int i = 0; i < 2; ++i)
        #pragma unroll
        for (int j = 0; j < 4; ++j) acc[i][j] = v8f{};

    stage(0, 0);
    ASYNC_WAIT();
    __syncthreads();

    int parity = 0;
    for (int kt = 0; kt < K; kt += BK) {
        int nxt = parity ^ 1;
        if (kt + BK < K) stage(kt + BK, nxt);   // overlap with compute below

        v16h af[2], bf[4];
        #pragma unroll
        for (int mf = 0; mf < 2; ++mf)
            af[mf] = load_a_frag(&sA[parity][(wm * 32 + mf * 16) * LDT], LDT, lane);
        #pragma unroll
        for (int nf = 0; nf < 4; ++nf)
            bf[nf] = load_b_frag(&sB[parity][(wn * 64 + nf * 16) * LDT], LDT, lane);
        #pragma unroll
        for (int mf = 0; mf < 2; ++mf)
            #pragma unroll
            for (int nf = 0; nf < 4; ++nf)
                acc[mf][nf] = __builtin_amdgcn_wmma_f32_16x16x32_f16(
                    false, af[mf], false, bf[nf], (short)0, acc[mf][nf],
                    false, false);

        ASYNC_WAIT();       // this wave's async writes into buf[nxt] landed
        __syncthreads();    // all waves: done reading cur, done writing nxt
        parity = nxt;
    }

    // epilogue: C-frag element r -> row m0+..+(lane[4]?8:0)+r, col = lane&15
    const int hi8 = (lane >> 4) << 3;
    const int nc  = lane & 15;
    #pragma unroll
    for (int mf = 0; mf < 2; ++mf) {
        #pragma unroll
        for (int nf = 0; nf < 4; ++nf) {
            int gmB = m0 + wm * 32 + mf * 16 + hi8;
            int gn  = n0 + wn * 64 + nf * 16 + nc;
            if constexpr (MODE == 0) {
                float bv = bias[gn];
                int which = gn >> 9;          // 0:q 1:k 2:v
                int oc = gn & (C_ - 1);
                #pragma unroll
                for (int r = 0; r < 8; ++r) {
                    int gm = gmB + r;
                    _Float16 val = (_Float16)(acc[mf][nf][r] + bv);
                    if (which == 0)      oq[(size_t)gm * C_ + oc] = val;
                    else if (which == 1) ok[(size_t)gm * C_ + oc] = val;
                    else {
                        int b = gm >> 12, n = gm & (N_ - 1);
                        ov[((size_t)b * C_ + oc) * N_ + n] = val;
                    }
                }
            } else if constexpr (MODE == 1) {
                #pragma unroll
                for (int r = 0; r < 8; ++r)
                    oq[(size_t)(gmB + r) * Nn + gn] =
                        (_Float16)(acc[mf][nf][r] * scale);
            } else { // MODE 3: final projection + residual, (b,c,n) f32
                float bv = bias[gn];
                #pragma unroll
                for (int r = 0; r < 8; ++r) {
                    int gm = gmB + r;
                    int b = gm >> 12, n = gm & (N_ - 1);
                    size_t idx = ((size_t)(b * C_ + gn)) * N_ + n;
                    of32[idx] = acc[mf][nf][r] + bv + resid[idx];
                }
            }
        }
    }
}

// ---------------------------------------------------------------------------
// Host launcher
// ---------------------------------------------------------------------------
extern "C" void kernel_launch(void* const* d_in, const int* in_sizes, int n_in,
                              void* d_out, int out_size, void* d_ws, size_t ws_size,
                              hipStream_t stream) {
    (void)in_sizes; (void)n_in; (void)out_size; (void)ws_size;
    const float* x     = (const float*)d_in[0];
    const float* gn_w  = (const float*)d_in[1];
    const float* gn_b  = (const float*)d_in[2];
    const float* qkv_w = (const float*)d_in[3];
    const float* qkv_b = (const float*)d_in[4];
    const float* out_w = (const float*)d_in[5];
    const float* out_b = (const float*)d_in[6];
    float* out = (float*)d_out;

    char* ws = (char*)d_ws;
    size_t off = 0;
    auto alloc = [&](size_t bytes) -> char* {
        char* p = ws + off;
        off += (bytes + 255) & ~(size_t)255;
        return p;
    };
    _Float16* h16    = (_Float16*)alloc((size_t)M_ * C_ * 2);
    _Float16* q16    = (_Float16*)alloc((size_t)M_ * C_ * 2);
    _Float16* k16    = (_Float16*)alloc((size_t)M_ * C_ * 2);
    _Float16* vt16   = (_Float16*)alloc((size_t)M_ * C_ * 2); // (b,c,n)
    _Float16* o16    = (_Float16*)alloc((size_t)M_ * C_ * 2); // (b,n,c)
    _Float16* qkvw16 = (_Float16*)alloc((size_t)3 * C_ * C_ * 2);
    _Float16* outw16 = (_Float16*)alloc((size_t)C_ * C_ * 2);
    _Float16* s16    = (_Float16*)alloc((size_t)N_ * N_ * 2);  // per-batch, reused
    float*    stats  = (float*)alloc((size_t)2 * B_ * G_ * 4);

    const float scale_s = 0.04419417382415922f; // 512^-0.5

    // 1. weight conversion
    cvt_f16<<<(3 * C_ * C_ + 255) / 256, 256, 0, stream>>>(qkv_w, qkvw16, 3 * C_ * C_);
    cvt_f16<<<(C_ * C_ + 255) / 256, 256, 0, stream>>>(out_w, outw16, C_ * C_);

    // 2. group norm
    gn_stats<<<B_ * G_, 256, 0, stream>>>(x, stats);
    gn_apply<<<B_ * C_, 256, 0, stream>>>(x, gn_w, gn_b, stats, h16);

    // 3. QKV projection: (16384 x 1536 x 512)
    gemm_wmma<0><<<dim3(3 * C_ / 128, M_ / 128), 256, 0, stream>>>(
        h16, qkvw16, 3 * C_, C_, q16, k16, vt16, nullptr, qkv_b, nullptr, 1.f);

    // 4. attention, batch-sequential (shared S buffer)
    for (int b = 0; b < B_; ++b) {
        size_t qoff = (size_t)b * N_ * C_;
        // S = scale * Q K^T   (4096 x 4096 x 512)
        gemm_wmma<1><<<dim3(N_ / 128, N_ / 128), 256, 0, stream>>>(
            q16 + qoff, k16 + qoff, N_, C_, s16, nullptr, nullptr, nullptr,
            nullptr, nullptr, scale_s);
        // softmax rows
        softmax_rows<<<N_, 256, 0, stream>>>(s16);
        // O = P V            (4096 x 512 x 4096), Bt = v^T (c,n)
        gemm_wmma<1><<<dim3(C_ / 128, N_ / 128), 256, 0, stream>>>(
            s16, vt16 + qoff, C_, N_, o16 + qoff, nullptr, nullptr, nullptr,
            nullptr, nullptr, 1.f);
    }

    // 5. output projection + bias + residual: (16384 x 512 x 512)
    gemm_wmma<3><<<dim3(C_ / 128, M_ / 128), 256, 0, stream>>>(
        o16, outw16, C_, C_, nullptr, nullptr, nullptr, out, out_b, x, 1.f);
}
